// PytorchSTFT_42271068127681
// MI455X (gfx1250) — matrix operand order repaired and checked by reading, not compiled, not run on hardware
//
#include <hip/hip_runtime.h>
#include <hip/hip_bf16.h>

// ---------------------------------------------------------------------------
// STFT magnitude for MI455X (gfx1250, wave32, WMMA).
//   out[b][c][t] = sqrt( (F_t . Bre_c)^2 + (F_t . Bim_c)^2 )
// GEMM: M=frames(1025/batch), N=channels(513 re + 513 im), K=1024.
// bf16 inputs, fp32 accumulate via v_wmma_f32_16x16x32_bf16.
// Per-wave tile: M=32 (two 16-row sub-tiles sharing B fragments) x N=16,
// with re+im accumulated simultaneously -> 4 WMMA per 8 x b128 loads.
// ---------------------------------------------------------------------------

#define N_FFT    1024
#define HOP      256
#define CUT      513          // output channels
#define NCH      1026         // basis rows (re + im)
#define NFRAMES  1025
#define NX       262144       // samples per batch
#define NBATCH   32
#define FT_TILES 33           // ceil(1025/32) frame tiles (32 frames/block)
#define CT_TILES 33           // ceil(513/16) channel tiles
#define SEG      (31 * HOP + N_FFT)   // 8960 samples staged per frame tile
#define OUT_PER_B (CUT * NFRAMES)     // 525825

typedef __bf16 v16bf __attribute__((ext_vector_type(16)));
typedef __bf16 v8bf  __attribute__((ext_vector_type(8)));
typedef float  v8f   __attribute__((ext_vector_type(8)));

union AFrag { v16bf v; v8bf h[2]; };
union Pack2 { __bf16 h[2]; unsigned u; };

// XOR-swizzle bits [6:4] of the LDS byte address with bits [11:9] so that the
// 512B-strided A rows hit different bank groups. Preserves 16B alignment.
__device__ __forceinline__ unsigned swz(unsigned byteoff) {
    return byteoff ^ (((byteoff >> 9) & 0x7u) << 4);
}

__device__ __forceinline__ int reflect_idx(int g) {
    if (g < 0)   g = -g;                 // left reflect
    if (g >= NX) g = 2 * NX - 2 - g;     // right reflect
    return g;
}

// --------------------------- basis fp32 -> bf16 ----------------------------
__global__ void stft_basis_to_bf16(const float* __restrict__ basis,
                                   __bf16* __restrict__ basis_bf, int n) {
    int i = blockIdx.x * blockDim.x + threadIdx.x;
    if (i < n) basis_bf[i] = (__bf16)basis[i];
}

// ------------------------------ main kernel --------------------------------
__global__ __launch_bounds__(256, 2)
void stft_mag_wmma_bf16(const float* __restrict__ x,
                        const __bf16* __restrict__ basis_bf,
                        float* __restrict__ out) {
    __shared__ __align__(16) unsigned char smem[SEG * 2];  // 17920 B bf16 frames

    const int t0    = blockIdx.x * 32;          // first frame of this tile
    const int b     = blockIdx.y;               // batch
    const int tid   = threadIdx.x;
    const int lane  = tid & 31;
    const int wave  = tid >> 5;

    // ---- stage x segment (reflect-padded) into LDS as bf16, swizzled ----
    // Two bf16 packed per ds_store_b32.
    const float* xb = x + (size_t)b * NX;
    const int seg_base = t0 * HOP - (N_FFT / 2);        // xp offset - pad
    for (int i = tid * 2; i < SEG; i += 512) {
        Pack2 p;
        p.h[0] = (__bf16)xb[reflect_idx(seg_base + i)];
        p.h[1] = (__bf16)xb[reflect_idx(seg_base + i + 1)];
        *(unsigned*)(smem + swz((unsigned)(i * 2))) = p.u;
    }
    __syncthreads();

    const int row  = lane & 15;   // A: frame row / B: column / C: N index
    const int half = lane >> 4;

    // ---- each wave walks channel tiles ct = wave, wave+8, ... < 33 ----
    for (int ct = wave; ct < CT_TILES; ct += 8) {
        const int c_base = ct * 16;
        const int c_re   = c_base + row;                       // <= 527
        int c_im         = c_base + CUT + row;                 // <= 1040
        if (c_im > NCH - 1) c_im = NCH - 1;                    // clamp (stores guarded)

        const __bf16* bre = basis_bf + (size_t)c_re * N_FFT + half * 16;
        const __bf16* bim = basis_bf + (size_t)c_im * N_FFT + half * 16;

        // prefetch next channel tile's basis rows into GL2 (global_prefetch_b8)
        if (ct + 8 < CT_TILES) {
            __builtin_prefetch(basis_bf + (size_t)(c_re + 128) * N_FFT, 0, 1);
            __builtin_prefetch(basis_bf + (size_t)(c_im + 128 <= NCH - 1 ?
                                                   c_im + 128 : NCH - 1) * N_FFT, 0, 1);
        }

        v8f acc_re_lo = {};
        v8f acc_im_lo = {};
        v8f acc_re_hi = {};
        v8f acc_im_hi = {};

        const unsigned arow_lo = (unsigned)(row * HOP);          // frames t0+row
        const unsigned arow_hi = (unsigned)((row + 16) * HOP);   // frames t0+16+row
        #pragma unroll 4
        for (int k0 = 0; k0 < N_FFT; k0 += 32) {
            // A fragments: contiguous k order (k = 16*half + s), same as B. lanes0-15: K 0-15,
            // lanes16-31: K 16-31 of row `row`.
            const unsigned kh = (unsigned)k0 + (unsigned)(half * 16);
            AFrag a_lo, a_hi;
            a_lo.h[0] = *(const v8bf*)(smem + swz((arow_lo + kh) * 2u));
            a_lo.h[1] = *(const v8bf*)(smem + swz((arow_lo + kh + 8u) * 2u));
            a_hi.h[0] = *(const v8bf*)(smem + swz((arow_hi + kh) * 2u));
            a_hi.h[1] = *(const v8bf*)(smem + swz((arow_hi + kh + 8u) * 2u));

            // B fragments: 32x16 layout, lane = column, K half by lane>>4;
            // 32B contiguous per lane. Shared by both M sub-tiles.
            v16bf bre_f = *(const v16bf*)(bre + k0);
            v16bf bim_f = *(const v16bf*)(bim + k0);

            acc_re_lo = __builtin_amdgcn_wmma_f32_16x16x32_bf16(
                false, a_lo.v, false, bre_f, (short)0, acc_re_lo, false, false);
            acc_im_lo = __builtin_amdgcn_wmma_f32_16x16x32_bf16(
                false, a_lo.v, false, bim_f, (short)0, acc_im_lo, false, false);
            acc_re_hi = __builtin_amdgcn_wmma_f32_16x16x32_bf16(
                false, a_hi.v, false, bre_f, (short)0, acc_re_hi, false, false);
            acc_im_hi = __builtin_amdgcn_wmma_f32_16x16x32_bf16(
                false, a_hi.v, false, bim_f, (short)0, acc_im_hi, false, false);
        }

        // ---- epilogue: magnitude + scatter per 16x16 f32 C layout ----
        // VGPR r, lanes0-15: M=r, N=lane; lanes16-31: M=r+8, N=lane-16.
        const int n_idx = lane & 15;
        const int c_out = c_base + n_idx;
        if (c_out < CUT) {
            float* ob = out + (size_t)b * OUT_PER_B + (size_t)c_out * NFRAMES;
            #pragma unroll
            for (int r = 0; r < 8; ++r) {
                const int m = r + half * 8;
                const int t_lo = t0 + m;
                const int t_hi = t0 + 16 + m;
                if (t_lo < NFRAMES) {
                    float re = acc_re_lo[r];
                    float im = acc_im_lo[r];
                    ob[t_lo] = __builtin_sqrtf(re * re + im * im);
                }
                if (t_hi < NFRAMES) {
                    float re = acc_re_hi[r];
                    float im = acc_im_hi[r];
                    ob[t_hi] = __builtin_sqrtf(re * re + im * im);
                }
            }
        }
    }
}

// ------------------------------- launcher ----------------------------------
extern "C" void kernel_launch(void* const* d_in, const int* in_sizes, int n_in,
                              void* d_out, int out_size, void* d_ws, size_t ws_size,
                              hipStream_t stream) {
    const float* x     = (const float*)d_in[0];   // (32, 262144) fp32
    const float* basis = (const float*)d_in[1];   // (1026, 1024) fp32
    float* out         = (float*)d_out;           // (32, 513, 1025) fp32
    __bf16* basis_bf   = (__bf16*)d_ws;           // 1026*1024*2 = 2,101,248 B

    const int nb = NCH * N_FFT;
    stft_basis_to_bf16<<<(nb + 255) / 256, 256, 0, stream>>>(basis, basis_bf, nb);

    dim3 grid(FT_TILES, NBATCH, 1);
    stft_mag_wmma_bf16<<<grid, 256, 0, stream>>>(x, basis_bf, out);
}
